// GATFusionBlock_54065048322201
// MI455X (gfx1250) — matrix operand order repaired and checked by reading, not compiled, not run on hardware
//
#include <hip/hip_runtime.h>
#include <hip/hip_bf16.h>
#include <cstddef>

typedef __attribute__((ext_vector_type(16))) __bf16 v16bf;
typedef __attribute__((ext_vector_type(8)))  float  v8f;
typedef int v4i_vec __attribute__((vector_size(16)));

#define BATCH 16
#define NSEQ  4096
#define DIM   384
#define KSEL  32
#define NEGV  -9e15f

#if defined(__gfx1250__) && __has_builtin(__builtin_amdgcn_global_load_async_to_lds_b128) && __has_builtin(__builtin_amdgcn_s_wait_asynccnt)
#define HAS_ASYNC_LDS 1
#else
#define HAS_ASYNC_LDS 0
#endif

static __device__ __forceinline__ __bf16 f2b(float f) { return (__bf16)f; }

static __device__ __forceinline__ v8f v8zero() {
  v8f z = {0.f,0.f,0.f,0.f,0.f,0.f,0.f,0.f};
  return z;
}

static __device__ __forceinline__ v8f wmma_bf16(v16bf a, v16bf b, v8f c) {
  return __builtin_amdgcn_wmma_f32_16x16x32_bf16(false, a, false, b, (short)0, c, false, false);
}

// 16-byte per-lane async copy global->LDS (ASYNCcnt) with portable fallback
static __device__ __forceinline__ void async_cp16(void* lds_dst, const void* gsrc) {
#if HAS_ASYNC_LDS
  __builtin_amdgcn_global_load_async_to_lds_b128(
      (__attribute__((address_space(1))) v4i_vec*)gsrc,
      (__attribute__((address_space(3))) v4i_vec*)lds_dst, 0, 0);
#else
  *(float4*)lds_dst = *(const float4*)gsrc;
#endif
}
static __device__ __forceinline__ void async_wait() {
#if HAS_ASYNC_LDS
  __builtin_amdgcn_s_wait_asynccnt(0);
#endif
}

// ---- fragment loaders (wave32 WMMA, contiguous k order shared by A and B) -------------
// A: 16x32 bf16. lane<16: row=lane,   K = k0+0..15
//                lane>=16: row=lane-16,K = k0+16..31
static __device__ __forceinline__ v16bf a_frag_lds(const __bf16* S, int ld, int row0, int k0) {
  int lane = threadIdx.x & 31;
  const __bf16* p = S + (row0 + (lane & 15)) * ld + k0 + ((lane >> 4) << 4);
  v16bf a;
#pragma unroll
  for (int i = 0; i < 16; ++i) a[i] = p[i];
  return a;
}
static __device__ __forceinline__ v16bf a_frag_lds_f32(const float* S, int ld, int row0, int k0) {
  int lane = threadIdx.x & 31;
  const float* p = S + (row0 + (lane & 15)) * ld + k0 + ((lane >> 4) << 4);
  v16bf a;
#pragma unroll
  for (int i = 0; i < 16; ++i) a[i] = f2b(p[i]);
  return a;
}
// B: 32x16. col = col0+(lane&15); lane<16: K=k0+0..15; lane>=16: K=k0+16..31
static __device__ __forceinline__ v16bf b_frag_g(const float* S, int ld, int k0, int col0) {
  int lane = threadIdx.x & 31;
  const float* p = S + (size_t)(k0 + ((lane >> 4) << 4)) * ld + col0 + (lane & 15);
  v16bf b;
#pragma unroll
  for (int i = 0; i < 16; ++i) b[i] = f2b(p[(size_t)i * ld]);
  return b;
}
static __device__ __forceinline__ v16bf b_frag_lds(const __bf16* S, int ld, int k0, int col0) {
  int lane = threadIdx.x & 31;
  const __bf16* p = S + (k0 + ((lane >> 4) << 4)) * ld + col0 + (lane & 15);
  v16bf b;
#pragma unroll
  for (int i = 0; i < 16; ++i) b[i] = p[i * ld];
  return b;
}
static __device__ __forceinline__ v16bf b_frag_lds_f32(const float* S, int ld, int k0, int col0) {
  int lane = threadIdx.x & 31;
  const float* p = S + (k0 + ((lane >> 4) << 4)) * ld + col0 + (lane & 15);
  v16bf b;
#pragma unroll
  for (int i = 0; i < 16; ++i) b[i] = f2b(p[i * ld]);
  return b;
}
// B from transposed row-major source T[n][k]: B[k][n] = T[n][k]
static __device__ __forceinline__ v16bf b_frag_trans_lds(const __bf16* T, int ld, int k0, int col0) {
  int lane = threadIdx.x & 31;
  const __bf16* p = T + (col0 + (lane & 15)) * ld + k0 + ((lane >> 4) << 4);
  v16bf b;
#pragma unroll
  for (int i = 0; i < 16; ++i) b[i] = p[i];
  return b;
}
static __device__ __forceinline__ v16bf b_frag_trans_g(const float* T, int ld, int k0, int col0) {
  int lane = threadIdx.x & 31;
  const float* p = T + (size_t)(col0 + (lane & 15)) * ld + k0 + ((lane >> 4) << 4);
  v16bf b;
#pragma unroll
  for (int i = 0; i < 16; ++i) b[i] = f2b(p[i]);
  return b;
}
// C/D: col = col0+(lane&15); row = row0 + v + 8*(lane>=16)
static __device__ __forceinline__ void store_cd_f32(float* D, int ld, int row0, int col0, v8f c) {
  int lane = threadIdx.x & 31;
  int col = col0 + (lane & 15);
  int rb = row0 + ((lane >> 4) << 3);
#pragma unroll
  for (int v = 0; v < 8; ++v) D[(rb + v) * ld + col] = c[v];
}
static __device__ __forceinline__ void store_cd_bf16(__bf16* D, int ld, int row0, int col0, v8f c) {
  int lane = threadIdx.x & 31;
  int col = col0 + (lane & 15);
  int rb = row0 + ((lane >> 4) << 3);
#pragma unroll
  for (int v = 0; v < 8; ++v) D[(rb + v) * ld + col] = f2b(c[v]);
}

static __device__ __forceinline__ float sigmoidf_(float x) { return 1.0f / (1.0f + __expf(-x)); }
static __device__ __forceinline__ float eluf_(float x) { return x > 0.0f ? x : __expf(x) - 1.0f; }

// ===========================================================================
// 1) masked means + have flags
__global__ void means_kernel(const float* __restrict__ x, const int* __restrict__ mask,
                             float* qp, float* qn, float* havep, float* haven) {
  int b = blockIdx.x, t = threadIdx.x;  // 384 threads
  const float* xb = x + (size_t)b * NSEQ * DIM;
  const int* mb = mask + (size_t)b * NSEQ;
  float ps = 0.f, ns = 0.f, pc = 0.f, nc = 0.f;
  for (int n = 0; n < NSEQ; ++n) {
    int mv = mb[n];
    float v = xb[(size_t)n * DIM + t];
    if (mv == 1) { ps += v; pc += 1.f; }
    else if (mv == -1) { ns += v; nc += 1.f; }
  }
  qp[b * DIM + t] = ps / fmaxf(pc, 1.f);
  qn[b * DIM + t] = ns / fmaxf(nc, 1.f);
  if (t == 0) { havep[b] = pc > 0.f ? 1.f : 0.f; haven[b] = nc > 0.f ? 1.f : 0.f; }
}

// 2) score vector v = Wk @ (Wq^T q) per (batch, branch)
__global__ void scorevec_kernel(const float* qp, const float* qn,
                                const float* sp_wq, const float* sp_wk,
                                const float* sn_wq, const float* sn_wk, float* vvec) {
  __shared__ float u[DIM];
  int blk = blockIdx.x, b = blk >> 1, br = blk & 1, t = threadIdx.x;
  const float* q  = (br ? qn : qp) + b * DIM;
  const float* Wq = br ? sn_wq : sp_wq;
  const float* Wk = br ? sn_wk : sp_wk;
  float s = 0.f;
  for (int i = 0; i < DIM; ++i) s += q[i] * Wq[i * DIM + t];
  u[t] = s;
  __syncthreads();
  float v = 0.f;
  for (int j = 0; j < DIM; ++j) v += Wk[t * DIM + j] * u[j];
  vvec[blk * DIM + t] = v;
}

// 3) sims: sigmoid(x . v / sqrt(d)) * have   (one wave per row)
__global__ void sim_kernel(const float* __restrict__ x, const float* vvec,
                           const float* havep, const float* haven, float* sim) {
  int wave = threadIdx.x >> 5, lane = threadIdx.x & 31;
  long r = (long)blockIdx.x * 8 + wave;       // (b*2+br)*N + n
  int n = (int)(r & (NSEQ - 1));
  int comb = (int)(r >> 12);
  int br = comb & 1, b = comb >> 1;
  const float* xr = x + ((size_t)b * NSEQ + n) * DIM;
  const float* vr = vvec + (size_t)comb * DIM;
  float s = 0.f;
  for (int c = lane; c < DIM; c += 32) s += xr[c] * vr[c];
  s += __shfl_xor(s, 16); s += __shfl_xor(s, 8); s += __shfl_xor(s, 4);
  s += __shfl_xor(s, 2);  s += __shfl_xor(s, 1);
  if (lane == 0) {
    float have = br ? haven[b] : havep[b];
    sim[(size_t)comb * NSEQ + n] = sigmoidf_(s * 0.05103103630798287f) * have;
  }
}

// 4) top-32 (iterative argmax) + validity flags
__global__ void topk_kernel(const float* sim, int* tk_idx, float* tk_real, float* kmaskv) {
  __shared__ float wv[NSEQ];
  __shared__ float rv[256];
  __shared__ int ri[256];
  __shared__ float vals[KSEL];
  __shared__ int idxs[KSEL];
  __shared__ float anyr;
  int blk = blockIdx.x, b = blk >> 1, br = blk & 1, tid = threadIdx.x;
  const float* s = sim + (size_t)blk * NSEQ;
  for (int i = tid; i < NSEQ; i += 256) wv[i] = s[i];
  __syncthreads();
  for (int k = 0; k < KSEL; ++k) {
    float bm = -1e30f; int bi = 0;
    for (int i = tid; i < NSEQ; i += 256) {
      float v = wv[i];
      if (v > bm) { bm = v; bi = i; }
    }
    rv[tid] = bm; ri[tid] = bi;
    __syncthreads();
    for (int st = 128; st > 0; st >>= 1) {
      if (tid < st) {
        float o = rv[tid + st]; int oi = ri[tid + st];
        if (o > rv[tid] || (o == rv[tid] && oi < ri[tid])) { rv[tid] = o; ri[tid] = oi; }
      }
      __syncthreads();
    }
    if (tid == 0) { vals[k] = rv[0]; idxs[k] = ri[0]; wv[ri[0]] = -1e30f; }
    __syncthreads();
  }
  if (tid == 0) {
    float a = 0.f;
    for (int k = 0; k < KSEL; ++k) if (vals[k] > 0.97f) a = 1.f;
    anyr = a;
  }
  __syncthreads();
  if (tid < KSEL) {
    bool real = vals[tid] > 0.97f;
    bool valid = real || (tid == 0 && anyr < 0.5f);
    tk_idx[blk * KSEL + tid] = idxs[tid];
    tk_real[blk * KSEL + tid] = real ? 1.f : 0.f;
    kmaskv[b * 64 + br * KSEL + tid] = valid ? 1.f : 0.f;
  }
}

// 5) per-(batch,branch) GAT: adj + 8 heads + output layer + pos-emb
__global__ void branch_kernel(const float* __restrict__ x, const float* pos_emb,
                              const float* adj_w,
                              const float* gp_W, const float* gp_a,
                              const float* gp_Wout, const float* gp_aout,
                              const float* gn_W, const float* gn_a,
                              const float* gn_Wout, const float* gn_aout,
                              const int* tk_idx, const float* tk_real,
                              const float* kmaskv, float* query) {
  extern __shared__ char smem[];
  __bf16* nodes = (__bf16*)smem;          // 32 x 392
  __bf16* buf1 = nodes + 32 * 392;        // 32 x 392 (p / Wh)
  __bf16* buf2 = buf1 + 32 * 392;         // 32 x 392 (elu(hp))
  float* adjm = (float*)(buf2 + 32 * 392);// 32 x 32
  float* attm = adjm + 32 * 32;           // 32 x 32
  float* svec = attm + 32 * 32;           // 64
  float* obuf = svec + 64;                // 32 x 392 f32 (C_out)

  int blk = blockIdx.x, b = blk >> 1, br = blk & 1;
  const float* gW    = br ? gn_W    : gp_W;
  const float* ga    = br ? gn_a    : gp_a;
  const float* gWout = br ? gn_Wout : gp_Wout;
  const float* gaout = br ? gn_aout : gp_aout;
  const int* idxp = tk_idx + blk * KSEL;
  const float* realp = tk_real + blk * KSEL;
  int tid = threadIdx.x, wave = tid >> 5;

  __builtin_prefetch(gW + tid * 16, 0, 0);     // warm L2 for head weights
  __builtin_prefetch(gWout + tid * 16, 0, 0);

  // gather nodes (zero where !real) -> bf16 LDS
  for (int e = tid; e < 32 * DIM; e += 256) {
    int r = e / DIM, c = e % DIM;
    float v = (realp[r] > 0.5f) ? x[((size_t)b * NSEQ + idxp[r]) * DIM + c] : 0.f;
    nodes[r * 392 + c] = f2b(v);
  }
  __syncthreads();

  // p = nodes @ adj_w
  for (int t = wave; t < 48; t += 8) {
    int mt = t / 24, nt = t % 24;
    v8f c = v8zero();
    for (int kk = 0; kk < 12; ++kk)
      c = wmma_bf16(a_frag_lds(nodes, 392, mt * 16, kk * 32),
                    b_frag_g(adj_w, DIM, kk * 32, nt * 16), c);
    store_cd_bf16(buf1, 392, mt * 16, nt * 16, c);
  }
  __syncthreads();

  // adj = sigmoid(p p^T / sqrt(d)), self-loops, validity mask
  if (wave < 4) {
    int mt = wave >> 1, nt = wave & 1;
    v8f c = v8zero();
    for (int kk = 0; kk < 12; ++kk)
      c = wmma_bf16(a_frag_lds(buf1, 392, mt * 16, kk * 32),
                    b_frag_trans_lds(buf1, 392, kk * 32, nt * 16), c);
    store_cd_f32(adjm, 32, mt * 16, nt * 16, c);
  }
  __syncthreads();
  for (int e = tid; e < 1024; e += 256) {
    int i = e >> 5, j = e & 31;
    float a_ = sigmoidf_(adjm[e] * 0.05103103630798287f);
    if (i == j) a_ = 1.f;
    float vi = kmaskv[b * 64 + br * KSEL + i];
    float vj = kmaskv[b * 64 + br * KSEL + j];
    adjm[e] = (vi > 0.5f && vj > 0.5f) ? a_ : 0.f;
  }
  __syncthreads();

  // 8 GAT heads; C_out accumulated in persistent registers (6 tiles / wave)
  v8f cacc[6];
#pragma unroll
  for (int i = 0; i < 6; ++i) cacc[i] = v8zero();

  for (int h = 0; h < 8; ++h) {
    const float* Wh = gW + (size_t)h * DIM * DIM;
    for (int t = wave; t < 48; t += 8) {  // Wh = nodes @ W_h
      int mt = t / 24, nt = t % 24;
      v8f c = v8zero();
      for (int kk = 0; kk < 12; ++kk)
        c = wmma_bf16(a_frag_lds(nodes, 392, mt * 16, kk * 32),
                      b_frag_g(Wh, DIM, kk * 32, nt * 16), c);
      store_cd_bf16(buf1, 392, mt * 16, nt * 16, c);
    }
    __syncthreads();
    const float* av = ga + h * 2 * DIM;
    if (tid < 64) {                        // s1/s2 matvecs
      int row = tid & 31, half = tid >> 5;
      const float* ap = av + half * DIM;
      const __bf16* wr = buf1 + row * 392;
      float s = 0.f;
      for (int d_ = 0; d_ < DIM; ++d_) s += (float)wr[d_] * ap[d_];
      svec[tid] = s;
    }
    __syncthreads();
    for (int e = tid; e < 1024; e += 256) {
      int i = e >> 5, j = e & 31;
      float ee = svec[i] + svec[32 + j];
      ee = ee > 0.f ? ee : 0.2f * ee;      // leaky_relu alpha=0.2
      attm[e] = adjm[e] > 0.5f ? ee : NEGV;
    }
    __syncthreads();
    if (tid < 32) {                        // row softmax over 32
      float m = -3.4e38f;
      for (int j = 0; j < 32; ++j) m = fmaxf(m, attm[tid * 32 + j]);
      float l = 0.f;
      for (int j = 0; j < 32; ++j) { float p = __expf(attm[tid * 32 + j] - m); attm[tid * 32 + j] = p; l += p; }
      float inv = 1.f / l;
      for (int j = 0; j < 32; ++j) attm[tid * 32 + j] *= inv;
    }
    __syncthreads();
    for (int t = wave; t < 48; t += 8) {   // hp = att @ Wh, elu -> buf2
      int mt = t / 24, nt = t % 24;
      v8f c = v8zero();
      c = wmma_bf16(a_frag_lds_f32(attm, 32, mt * 16, 0),
                    b_frag_lds(buf1, 392, 0, nt * 16), c);
      int lane = tid & 31, col = nt * 16 + (lane & 15), rb = mt * 16 + ((lane >> 4) << 3);
#pragma unroll
      for (int v = 0; v < 8; ++v) buf2[(rb + v) * 392 + col] = f2b(eluf_(c[v]));
    }
    __syncthreads();
    const float* Wo_h = gWout + (size_t)(h * DIM) * DIM;
#pragma unroll
    for (int i = 0; i < 6; ++i) {          // C_out += elu(hp) @ Wout[h-block]
      int t = wave * 6 + i, mt = t / 24, nt = t % 24;
      for (int kk = 0; kk < 12; ++kk)
        cacc[i] = wmma_bf16(a_frag_lds(buf2, 392, mt * 16, kk * 32),
                            b_frag_g(Wo_h, DIM, kk * 32, nt * 16), cacc[i]);
    }
    __syncthreads();
  }
#pragma unroll
  for (int i = 0; i < 6; ++i) {
    int t = wave * 6 + i, mt = t / 24, nt = t % 24;
    store_cd_f32(obuf, 392, mt * 16, nt * 16, cacc[i]);
  }
  __syncthreads();

  // output GAT layer
  if (tid < 64) {
    int row = tid & 31, half = tid >> 5;
    const float* ap = gaout + half * DIM;
    const float* wr = obuf + row * 392;
    float s = 0.f;
    for (int d_ = 0; d_ < DIM; ++d_) s += wr[d_] * ap[d_];
    svec[tid] = s;
  }
  __syncthreads();
  for (int e = tid; e < 1024; e += 256) {
    int i = e >> 5, j = e & 31;
    float ee = svec[i] + svec[32 + j];
    ee = ee > 0.f ? ee : 0.2f * ee;
    attm[e] = adjm[e] > 0.5f ? ee : NEGV;
  }
  __syncthreads();
  if (tid < 32) {
    float m = -3.4e38f;
    for (int j = 0; j < 32; ++j) m = fmaxf(m, attm[tid * 32 + j]);
    float l = 0.f;
    for (int j = 0; j < 32; ++j) { float p = __expf(attm[tid * 32 + j] - m); attm[tid * 32 + j] = p; l += p; }
    float inv = 1.f / l;
    for (int j = 0; j < 32; ++j) attm[tid * 32 + j] *= inv;
  }
  __syncthreads();
  for (int t = wave; t < 48; t += 8) {     // elu(att @ C_out) + pe -> query
    int mt = t / 24, nt = t % 24;
    v8f c = v8zero();
    c = wmma_bf16(a_frag_lds_f32(attm, 32, mt * 16, 0),
                  b_frag_lds_f32(obuf, 392, 0, nt * 16), c);
    int lane = tid & 31, col = nt * 16 + (lane & 15), rb = mt * 16 + ((lane >> 4) << 3);
#pragma unroll
    for (int v = 0; v < 8; ++v) {
      int row = rb + v;
      float val = eluf_(c[v]);
      if (realp[row] > 0.5f) val += pos_emb[(size_t)idxp[row] * DIM + col];
      query[((size_t)b * 64 + br * KSEL + row) * DIM + col] = val;
    }
  }
}

// 6) i2q cross-attention: query[64] attends feat[4096], flash-style per (b, head)
__global__ void i2q_kernel(const float* __restrict__ x, const float* query,
                           const float* wq, const float* wk, const float* wv,
                           float* attout) {
  extern __shared__ char smem[];
  float* featt = (float*)smem;            // 64 x 392 f32 (async-staged)
  __bf16* qh = (__bf16*)(featt + 64 * 392);  // 64 x 104
  __bf16* kbuf = qh + 64 * 104;           // 64 x 104
  __bf16* vbuf = kbuf + 64 * 104;         // 64 x 104
  float* sbuf = (float*)(vbuf + 64 * 104);// 64 x 68
  float* acc = sbuf + 64 * 68;            // 64 x 96
  float* mrow = acc + 64 * 96;            // 64
  float* lrow = mrow + 64;                // 64
  int b = blockIdx.x >> 2, h = blockIdx.x & 3;
  int tid = threadIdx.x, wave = tid >> 5;

  __builtin_prefetch(wk + tid * 16, 0, 0);
  __builtin_prefetch(wv + tid * 16, 0, 0);

  // stage query tile via async DMA (f32, 16B chunks: 64 rows x 96 chunks)
  {
    const float* qs = query + (size_t)b * 64 * DIM;
    for (int e = tid; e < 64 * 96; e += 256) {
      int r = e / 96, o = (e % 96) * 4;
      async_cp16(&featt[r * 392 + o], &qs[(size_t)r * DIM + o]);
    }
    async_wait();
  }
  __syncthreads();
  for (int t = wave; t < 24; t += 8) {    // q_h = query @ wq[:, h]
    int mt = t / 6, nt = t % 6;
    v8f c = v8zero();
    for (int kk = 0; kk < 12; ++kk)
      c = wmma_bf16(a_frag_lds_f32(featt, 392, mt * 16, kk * 32),
                    b_frag_g(wq, DIM, kk * 32, h * 96 + nt * 16), c);
    store_cd_bf16(qh, 104, mt * 16, nt * 16, c);
  }
  for (int e = tid; e < 64 * 96; e += 256) acc[e] = 0.f;
  if (tid < 64) { mrow[tid] = -3.4e38f; lrow[tid] = 0.f; }
  __syncthreads();

  const float rs = 0.10206207261596575f;  // 1/sqrt(96)
  for (int kt = 0; kt < NSEQ / 64; ++kt) {
    const float* xs = x + ((size_t)b * NSEQ + kt * 64) * DIM;
    for (int e = tid; e < 64 * 96; e += 256) {   // async stage feat tile
      int r = e / 96, o = (e % 96) * 4;
      async_cp16(&featt[r * 392 + o], &xs[(size_t)r * DIM + o]);
    }
    async_wait();
    __syncthreads();
    for (int t = wave; t < 48; t += 8) {  // k_t / v_t head slices
      int which = t / 24, tt = t % 24, mt = tt / 6, nt = tt % 6;
      const float* W = which ? wv : wk;
      v8f c = v8zero();
      for (int kk = 0; kk < 12; ++kk)
        c = wmma_bf16(a_frag_lds_f32(featt, 392, mt * 16, kk * 32),
                      b_frag_g(W, DIM, kk * 32, h * 96 + nt * 16), c);
      store_cd_bf16(which ? vbuf : kbuf, 104, mt * 16, nt * 16, c);
    }
    __syncthreads();
    for (int t = wave; t < 16; t += 8) {  // S = q_h @ k_t^T
      int mt = t / 4, nt = t % 4;
      v8f c = v8zero();
      for (int kk = 0; kk < 3; ++kk)
        c = wmma_bf16(a_frag_lds(qh, 104, mt * 16, kk * 32),
                      b_frag_trans_lds(kbuf, 104, kk * 32, nt * 16), c);
      store_cd_f32(sbuf, 68, mt * 16, nt * 16, c);
    }
    __syncthreads();
    if (tid < 64) {                        // online softmax update
      int r = tid;
      float m = mrow[r], tmax = -3.4e38f;
      for (int j = 0; j < 64; ++j) { float v = sbuf[r * 68 + j] * rs; sbuf[r * 68 + j] = v; tmax = fmaxf(tmax, v); }
      float mn = fmaxf(m, tmax);
      float corr = __expf(m - mn);
      float ps = 0.f;
      for (int j = 0; j < 64; ++j) { float p = __expf(sbuf[r * 68 + j] - mn); sbuf[r * 68 + j] = p; ps += p; }
      lrow[r] = lrow[r] * corr + ps; mrow[r] = mn;
      for (int c_ = 0; c_ < 96; ++c_) acc[r * 96 + c_] *= corr;
    }
    __syncthreads();
    for (int t = wave; t < 24; t += 8) {  // acc += P @ V_t
      int mt = t / 6, nt = t % 6;
      v8f c = v8zero();
      for (int kk = 0; kk < 2; ++kk)
        c = wmma_bf16(a_frag_lds_f32(sbuf, 68, mt * 16, kk * 32),
                      b_frag_lds(vbuf, 104, kk * 32, nt * 16), c);
      int lane = tid & 31, col = nt * 16 + (lane & 15), rb = mt * 16 + ((lane >> 4) << 3);
#pragma unroll
      for (int v = 0; v < 8; ++v) acc[(rb + v) * 96 + col] += c[v];
    }
    __syncthreads();
  }
  for (int e = tid; e < 64 * 96; e += 256) {
    int r = e / 96, c = e % 96;
    attout[((size_t)b * 64 + r) * DIM + h * 96 + c] = acc[e] / lrow[r];
  }
}

// 7) [64,384]@[384,384] (+bias) per batch
__global__ void proj64_kernel(const float* X, const float* W, const float* bias, float* Y) {
  extern __shared__ char smem[];
  float* xb = (float*)smem;               // 64 x 392 f32 (async-staged)
  int b = blockIdx.x, tid = threadIdx.x, wave = tid >> 5;
  const float* Xb = X + (size_t)b * 64 * DIM;
  __builtin_prefetch(W + tid * 16, 0, 0);
  for (int e = tid; e < 64 * 96; e += 256) {
    int r = e / 96, o = (e % 96) * 4;
    async_cp16(&xb[r * 392 + o], &Xb[(size_t)r * DIM + o]);
  }
  async_wait();
  __syncthreads();
  for (int t = wave; t < 96; t += 8) {
    int mt = t / 24, nt = t % 24;
    v8f c = v8zero();
    for (int kk = 0; kk < 12; ++kk)
      c = wmma_bf16(a_frag_lds_f32(xb, 392, mt * 16, kk * 32),
                    b_frag_g(W, DIM, kk * 32, nt * 16), c);
    int lane = tid & 31, col = nt * 16 + (lane & 15), rb = mt * 16 + ((lane >> 4) << 3);
#pragma unroll
    for (int v = 0; v < 8; ++v) {
      float val = c[v];
      if (bias) val += bias[col];
      Y[((size_t)b * 64 + rb + v) * DIM + col] = val;
    }
  }
}

// 8) q2i cross-attention + output projection, fused per 64-row tile of feat
__global__ void q2i_kernel(const float* __restrict__ x, const float* k2, const float* v2,
                           const float* wq, const float* wo, const float* bo,
                           const float* kmaskv, float* out) {
  extern __shared__ char smem[];
  float* featt = (float*)smem;            // 64 x 392 f32 (async-staged)
  __bf16* qbuf = (__bf16*)(featt + 64 * 392); // 64 x 392
  __bf16* obuf = qbuf + 64 * 392;         // 64 x 392
  float* sbuf = (float*)(obuf + 64 * 392);// 64 x 68
  int b = blockIdx.x / (NSEQ / 64);
  int n0 = (blockIdx.x % (NSEQ / 64)) * 64;
  int tid = threadIdx.x, wave = tid >> 5;
  const float* xs = x + ((size_t)b * NSEQ + n0) * DIM;
  __builtin_prefetch(wq + tid * 16, 0, 0);
  __builtin_prefetch(wo + tid * 16, 0, 0);
  for (int e = tid; e < 64 * 96; e += 256) {   // async stage feat tile
    int r = e / 96, o = (e % 96) * 4;
    async_cp16(&featt[r * 392 + o], &xs[(size_t)r * DIM + o]);
  }
  async_wait();
  __syncthreads();
  for (int t = wave; t < 96; t += 8) {    // Q = feat @ wq
    int mt = t / 24, nt = t % 24;
    v8f c = v8zero();
    for (int kk = 0; kk < 12; ++kk)
      c = wmma_bf16(a_frag_lds_f32(featt, 392, mt * 16, kk * 32),
                    b_frag_g(wq, DIM, kk * 32, nt * 16), c);
    store_cd_bf16(qbuf, 392, mt * 16, nt * 16, c);
  }
  __syncthreads();
  const float rs = 0.10206207261596575f;
  const float* k2b = k2 + (size_t)b * 64 * DIM;
  const float* v2b = v2 + (size_t)b * 64 * DIM;
  for (int h = 0; h < 4; ++h) {
    for (int t = wave; t < 16; t += 8) {  // S = Q_h @ k2_h^T
      int mt = t / 4, nt = t % 4;
      v8f c = v8zero();
      for (int kk = 0; kk < 3; ++kk)
        c = wmma_bf16(a_frag_lds(qbuf, 392, mt * 16, h * 96 + kk * 32),
                      b_frag_trans_g(k2b, DIM, h * 96 + kk * 32, nt * 16), c);
      store_cd_f32(sbuf, 68, mt * 16, nt * 16, c);
    }
    __syncthreads();
    if (tid < 64) {                        // masked softmax over 64 keys
      int r = tid;
      float m = -3.4e38f;
      for (int j = 0; j < 64; ++j) {
        float v = sbuf[r * 68 + j] * rs;
        if (kmaskv[b * 64 + j] < 0.5f) v = NEGV;
        sbuf[r * 68 + j] = v; m = fmaxf(m, v);
      }
      float l = 0.f;
      for (int j = 0; j < 64; ++j) { float p = __expf(sbuf[r * 68 + j] - m); sbuf[r * 68 + j] = p; l += p; }
      float inv = 1.f / l;
      for (int j = 0; j < 64; ++j) sbuf[r * 68 + j] *= inv;
    }
    __syncthreads();
    for (int t = wave; t < 24; t += 8) {  // O_h = att @ v2_h -> obuf cols h*96..
      int mt = t / 6, nt = t % 6;
      v8f c = v8zero();
      for (int kk = 0; kk < 2; ++kk)
        c = wmma_bf16(a_frag_lds_f32(sbuf, 68, mt * 16, kk * 32),
                      b_frag_g(v2b, DIM, kk * 32, h * 96 + nt * 16), c);
      store_cd_bf16(obuf, 392, mt * 16, h * 96 + nt * 16, c);
    }
    __syncthreads();
  }
  for (int t = wave; t < 96; t += 8) {    // out = O @ wo + bo
    int mt = t / 24, nt = t % 24;
    v8f c = v8zero();
    for (int kk = 0; kk < 12; ++kk)
      c = wmma_bf16(a_frag_lds(obuf, 392, mt * 16, kk * 32),
                    b_frag_g(wo, DIM, kk * 32, nt * 16), c);
    int lane = tid & 31, col = nt * 16 + (lane & 15), rb = mt * 16 + ((lane >> 4) << 3);
#pragma unroll
    for (int v = 0; v < 8; ++v)
      out[((size_t)b * NSEQ + n0 + rb + v) * DIM + col] = c[v] + bo[col];
  }
}

// ===========================================================================
extern "C" void kernel_launch(void* const* d_in, const int* in_sizes, int n_in,
                              void* d_out, int out_size, void* d_ws, size_t ws_size,
                              hipStream_t stream) {
  const float* x       = (const float*)d_in[0];
  const int*   mask    = (const int*)d_in[1];
  const float* pos_emb = (const float*)d_in[2];
  const float* sp_wq   = (const float*)d_in[3];
  const float* sp_wk   = (const float*)d_in[4];
  const float* sn_wq   = (const float*)d_in[5];
  const float* sn_wk   = (const float*)d_in[6];
  const float* adj_w   = (const float*)d_in[7];
  const float* gp_W    = (const float*)d_in[8];
  const float* gp_a    = (const float*)d_in[9];
  const float* gp_Wout = (const float*)d_in[10];
  const float* gp_aout = (const float*)d_in[11];
  const float* gn_W    = (const float*)d_in[12];
  const float* gn_a    = (const float*)d_in[13];
  const float* gn_Wout = (const float*)d_in[14];
  const float* gn_aout = (const float*)d_in[15];
  const float* i2q_wq  = (const float*)d_in[16];
  const float* i2q_wk  = (const float*)d_in[17];
  const float* i2q_wv  = (const float*)d_in[18];
  const float* i2q_wo  = (const float*)d_in[19];
  const float* i2q_bo  = (const float*)d_in[20];
  const float* q2i_wq  = (const float*)d_in[21];
  const float* q2i_wk  = (const float*)d_in[22];
  const float* q2i_wv  = (const float*)d_in[23];
  const float* q2i_wo  = (const float*)d_in[24];
  const float* q2i_bo  = (const float*)d_in[25];
  float* out = (float*)d_out;

  float* w = (float*)d_ws;
  float* qp = w;      w += BATCH * DIM;
  float* qn = w;      w += BATCH * DIM;
  float* havep = w;   w += BATCH;
  float* haven = w;   w += BATCH;
  float* vvec = w;    w += BATCH * 2 * DIM;
  float* sim = w;     w += BATCH * 2 * NSEQ;
  int*   tk_idx = (int*)w; w += BATCH * 2 * KSEL;
  float* tk_real = w; w += BATCH * 2 * KSEL;
  float* kmaskv = w;  w += BATCH * 64;
  float* query = w;   w += BATCH * 64 * DIM;
  float* attout = w;  w += BATCH * 64 * DIM;
  float* query2 = w;  w += BATCH * 64 * DIM;
  float* k2 = w;      w += BATCH * 64 * DIM;
  float* v2 = w;      w += BATCH * 64 * DIM;

  means_kernel<<<BATCH, DIM, 0, stream>>>(x, mask, qp, qn, havep, haven);
  scorevec_kernel<<<BATCH * 2, DIM, 0, stream>>>(qp, qn, sp_wq, sp_wk, sn_wq, sn_wk, vvec);
  sim_kernel<<<BATCH * 2 * NSEQ / 8, 256, 0, stream>>>(x, vvec, havep, haven, sim);
  topk_kernel<<<BATCH * 2, 256, 0, stream>>>(sim, tk_idx, tk_real, kmaskv);

  size_t lds_branch = 3 * (32 * 392 * sizeof(__bf16)) + 2 * (32 * 32 * sizeof(float)) +
                      64 * sizeof(float) + 32 * 392 * sizeof(float);
  branch_kernel<<<BATCH * 2, 256, lds_branch, stream>>>(
      x, pos_emb, adj_w, gp_W, gp_a, gp_Wout, gp_aout,
      gn_W, gn_a, gn_Wout, gn_aout, tk_idx, tk_real, kmaskv, query);

  size_t lds_i2q = 64 * 392 * sizeof(float) + 3 * (64 * 104 * sizeof(__bf16)) +
                   64 * 68 * sizeof(float) + 64 * 96 * sizeof(float) + 2 * 64 * sizeof(float);
  i2q_kernel<<<BATCH * 4, 256, lds_i2q, stream>>>(x, query, i2q_wq, i2q_wk, i2q_wv, attout);

  size_t lds_proj = 64 * 392 * sizeof(float);
  proj64_kernel<<<BATCH, 256, lds_proj, stream>>>(attout, i2q_wo, i2q_bo, query2);
  proj64_kernel<<<BATCH, 256, lds_proj, stream>>>(query2, q2i_wk, nullptr, k2);
  proj64_kernel<<<BATCH, 256, lds_proj, stream>>>(query2, q2i_wv, nullptr, v2);

  size_t lds_q2i = 64 * 392 * sizeof(float) + 2 * (64 * 392 * sizeof(__bf16)) +
                   64 * 68 * sizeof(float);
  q2i_kernel<<<BATCH * (NSEQ / 64), 256, lds_q2i, stream>>>(
      x, k2, v2, q2i_wq, q2i_wo, q2i_bo, kmaskv, out);
}
